// SS2D_HIGHFREQ_5712306504328
// MI455X (gfx1250) — matrix operand order repaired and checked by reading, not compiled, not run on hardware
//
#include <hip/hip_runtime.h>
#include <hip/hip_bf16.h>
#include <math.h>

#define BATCH  2
#define HH     48
#define WW     48
#define DMODEL 96
#define DSTATE 16
#define DINNER 192
#define DTRANK 6
#define KDIR   6
#define LL     (HH * WW)        // 2304
#define BL     (BATCH * LL)     // 4608
#define SCH    32               // scan chunks
#define SCLEN  (LL / SCH)       // 72 steps per chunk

typedef __attribute__((ext_vector_type(16))) _Float16 v16h;
typedef __attribute__((ext_vector_type(8)))  float    v8f;

__device__ inline float fast_rcp(float x) { return __builtin_amdgcn_rcpf(x); }
__device__ inline float silu_f(float v)   { return v * fast_rcp(1.f + __expf(-v)); }

// ---------------------------------------------------------------------------
// Wave-level 16x16 output tile GEMM using V_WMMA_F32_16X16X32_F16.
// A: row-major (16 x KTOT) fp32 at Arow, leading dim lda.
// B: weights W[n][k] row-major (B[k][n] = Wrow[n*ldw + k]).
// nvalid: valid columns in this 16-wide N tile (zero-pad rest).
// Per-lane gathers follow CDNA5 ISA 7.12.2 wave32 layouts.
// ---------------------------------------------------------------------------
template<int KTOT>
__device__ inline v8f wave_gemm_tile(const float* __restrict__ Arow, int lda,
                                     const float* __restrict__ Wrow, int ldw,
                                     int nvalid)
{
  const int lane  = threadIdx.x & 31;
  const int lhalf = lane >> 4;      // 0: lanes 0-15, 1: lanes 16-31
  const int arow  = lane & 15;      // A-matrix row M
  const int ncol  = lane & 15;      // B/C/D column N
  v8f acc = {0.f, 0.f, 0.f, 0.f, 0.f, 0.f, 0.f, 0.f};
  const float* aptr = Arow + (long)arow * lda;
  const bool  nok  = (ncol < nvalid);
  const float* bptr = Wrow + (long)(nok ? ncol : 0) * ldw;
  #pragma unroll
  for (int kb = 0; kb < KTOT; kb += 32) {
    v16h a, b;
    const int abase = kb + lhalf * 8;     // A: K = kb + {0..7,16..23} (+8 hi lanes)
    #pragma unroll
    for (int i = 0; i < 8; ++i) {
      a[i]     = (_Float16)aptr[abase + i];
      a[i + 8] = (_Float16)aptr[abase + 16 + i];
    }
    const int bbase = kb + lhalf * 8;     // B: K = kb + {0..7,16..23} (+8 hi lanes)
    #pragma unroll
    for (int i = 0; i < 16; ++i) {
      float v = nok ? bptr[bbase + (i & 7) + ((i >> 3) << 4)] : 0.f;
      b[i] = (_Float16)v;
    }
    acc = __builtin_amdgcn_wmma_f32_16x16x32_f16(
        false, a, false, b, (short)0, acc, false, false);
  }
  return acc;
}

// ---------------------------------------------------------------------------
// 1) Input projection GEMM; optional SiLU (for z -> zg).
// ---------------------------------------------------------------------------
__global__ void k_inproj(const float* __restrict__ X,   // (BL, 96)
                         const float* __restrict__ Wp,  // (384, 96)
                         float* __restrict__ Outp,      // (BL, 192)
                         int n_off, int act_silu)
{
  const int NT = DINNER / 16;  // 12
  const int tile  = blockIdx.x;
  const int mtile = tile / NT;
  const int ntile = tile % NT;
  const float* Arow = X + (long)mtile * 16 * DMODEL;
  const float* Wrow = Wp + (long)(n_off + ntile * 16) * DMODEL;
  v8f acc = wave_gemm_tile<DMODEL>(Arow, DMODEL, Wrow, DMODEL, 16);
  const int lane = threadIdx.x & 31;
  const int lhalf = lane >> 4, ncol = lane & 15;
  #pragma unroll
  for (int r = 0; r < 8; ++r) {
    const int m = mtile * 16 + r + lhalf * 8;
    const int c = ntile * 16 + ncol;
    float v = acc[r];
    if (act_silu) v = silu_f(v);
    Outp[(long)m * DINNER + c] = v;
  }
}

// ---------------------------------------------------------------------------
// 2) Depthwise 3x3 conv (pad 1) + SiLU, channel-last.
// ---------------------------------------------------------------------------
__global__ void k_conv3x3_silu(const float* __restrict__ In,  // (B,H,W,DI)
                               const float* __restrict__ Wc,  // (DI,1,3,3)
                               const float* __restrict__ bc,  // (DI)
                               float* __restrict__ Outp)
{
  const long idx = (long)blockIdx.x * blockDim.x + threadIdx.x;
  if (idx >= (long)BL * DINNER) return;
  const int d = (int)(idx % DINNER);
  const long m = idx / DINNER;
  const int w = (int)(m % WW);
  const int h = (int)((m / WW) % HH);
  const int b = (int)(m / LL);
  float acc = bc[d];
  #pragma unroll
  for (int di = -1; di <= 1; ++di) {
    const int hh = h + di;
    if (hh < 0 || hh >= HH) continue;
    #pragma unroll
    for (int dj = -1; dj <= 1; ++dj) {
      const int ww2 = w + dj;
      if (ww2 < 0 || ww2 >= WW) continue;
      acc += Wc[d * 9 + (di + 1) * 3 + (dj + 1)] *
             In[(((long)b * HH + hh) * WW + ww2) * DINNER + d];
    }
  }
  Outp[idx] = silu_f(acc);
}

// ---------------------------------------------------------------------------
// 3) Build the 6 directional sequences xs (B,6,L,DI), channel-last.
// ---------------------------------------------------------------------------
__global__ void k_build_xs(const float* __restrict__ xc,
                           const float* __restrict__ yc,
                           float* __restrict__ xs)
{
  const long idx = (long)blockIdx.x * blockDim.x + threadIdx.x;
  if (idx >= (long)BATCH * KDIR * LL * DINNER) return;
  const int d = (int)(idx % DINNER);
  long r0 = idx / DINNER;
  const int l = (int)(r0 % LL);
  long r1 = r0 / LL;
  const int k = (int)(r1 % KDIR);
  const int b = (int)(r1 / KDIR);
  const int kk = k % 3;
  const int ls = (k >= 3) ? (LL - 1 - l) : l;
  float v;
  if (kk == 0) {
    v = xc[((long)b * LL + ls) * DINNER + d];
  } else {
    const int h  = ls % HH;
    const int wq = ls / HH;
    const int w  = (kk == 1) ? wq : ((h + wq) % WW);
    v = yc[((long)b * LL + h * WW + w) * DINNER + d];
  }
  xs[idx] = v;
}

// ---------------------------------------------------------------------------
// 4) x_proj GEMM per direction: N padded 38->48; route c -> dtr/Bs/Cs.
// ---------------------------------------------------------------------------
__global__ void k_xdbl(const float* __restrict__ xs,   // (B,6,L,DI)
                       const float* __restrict__ Wxp,  // (6,38,192)
                       float* __restrict__ dtr,        // (B,6,L,6)
                       float* __restrict__ Bsv,        // (B,6,L,16)
                       float* __restrict__ Csv)        // (B,6,L,16)
{
  const int MT = BL / 16;  // 288
  const int NT = 3;
  const int tile = blockIdx.x;
  const int k = tile / (MT * NT);
  const int rem = tile % (MT * NT);
  const int mtile = rem / NT;
  const int ntile = rem % NT;
  const int m0 = mtile * 16;
  const int b  = m0 / LL;
  const int l0 = m0 % LL;
  const float* Arow = xs + (((long)(b * KDIR + k)) * LL + l0) * DINNER;
  const int nvalid = (ntile == 2) ? (38 - 32) : 16;
  const float* Wrow = Wxp + ((long)k * 38 + ntile * 16) * DINNER;
  v8f acc = wave_gemm_tile<DINNER>(Arow, DINNER, Wrow, DINNER, nvalid);
  const int lane = threadIdx.x & 31;
  const int lhalf = lane >> 4, ncol = lane & 15;
  const int c = ntile * 16 + ncol;
  #pragma unroll
  for (int r = 0; r < 8; ++r) {
    const int l = l0 + r + lhalf * 8;
    const long base = ((long)(b * KDIR + k)) * LL + l;
    const float v = acc[r];
    if (c < DTRANK)                    dtr[base * DTRANK + c] = v;
    else if (c < DTRANK + DSTATE)      Bsv[base * DSTATE + (c - DTRANK)] = v;
    else if (c < DTRANK + 2 * DSTATE)  Csv[base * DSTATE + (c - DTRANK - DSTATE)] = v;
  }
}

// ---------------------------------------------------------------------------
// 5) delta = softplus(dtr @ dt_w^T + dt_b), (B,6,L,DI).
// ---------------------------------------------------------------------------
__global__ void k_delta(const float* __restrict__ dtr,  // (B,6,L,6)
                        const float* __restrict__ dtw,  // (6,192,6)
                        const float* __restrict__ dtb,  // (6,192)
                        float* __restrict__ delta)
{
  const long idx = (long)blockIdx.x * blockDim.x + threadIdx.x;
  if (idx >= (long)BATCH * KDIR * LL * DINNER) return;
  const int d = (int)(idx % DINNER);
  long r0 = idx / DINNER;
  const int l = (int)(r0 % LL);
  long r1 = r0 / LL;
  const int k = (int)(r1 % KDIR);
  const int b = (int)(r1 / KDIR);
  const long base = ((long)(b * KDIR + k)) * LL + l;
  float s = dtb[k * DINNER + d];
  #pragma unroll
  for (int r = 0; r < DTRANK; ++r)
    s += dtr[base * DTRANK + r] * dtw[((long)k * DINNER + d) * DTRANK + r];
  // native-exp/log softplus (v_exp_f32 / v_log_f32), overflow-guarded
  delta[idx] = (s > 20.f) ? s : __logf(1.f + __expf(s));
}

// ---------------------------------------------------------------------------
// 6a) Scan phase A: per-chunk transition product and partial state (h0 = 0).
//     h[t] = a[t]*h[t-1] + b[t],  a = exp(dl*A),  b = dl*u*B.
//     Per chunk: Aprod = prod(a),  Hpart = chunk state starting from 0.
// ---------------------------------------------------------------------------
__global__ void k_scan_chunks(const float* __restrict__ delta,
                              const float* __restrict__ xs,
                              const float* __restrict__ Bsv,
                              const float* __restrict__ A_logs, // (6*192,16)
                              float* __restrict__ Aprod,        // (B,6,SCH,DI,16)
                              float* __restrict__ Hpart)        // (B,6,SCH,DI,16)
{
  const int blk = blockIdx.x;       // bk*SCH + c
  const int c  = blk % SCH;
  const int bk = blk / SCH;
  const int k  = bk % KDIR;
  const int d  = threadIdx.x;
  float Arow[DSTATE], ap[DSTATE], hp[DSTATE];
  #pragma unroll
  for (int n = 0; n < DSTATE; ++n) {
    Arow[n] = -__expf(A_logs[((long)k * DINNER + d) * DSTATE + n]);
    ap[n] = 1.f;
    hp[n] = 0.f;
  }
  const int t0 = c * SCLEN;
  const float* dp = delta + (long)bk * LL * DINNER;
  const float* up = xs    + (long)bk * LL * DINNER;
  const float* Bp = Bsv   + (long)bk * LL * DSTATE;
  for (int tt = 0; tt < SCLEN; ++tt) {
    const int t = t0 + tt;
    const float dl = dp[(long)t * DINNER + d];
    const float u  = up[(long)t * DINNER + d];
    const float du = dl * u;
    #pragma unroll
    for (int n = 0; n < DSTATE; ++n) {
      const float e = __expf(dl * Arow[n]);
      ap[n] *= e;
      hp[n] = e * hp[n] + du * Bp[(long)t * DSTATE + n];
    }
  }
  const long o = (((long)bk * SCH + c) * DINNER + d) * DSTATE;
  #pragma unroll
  for (int n = 0; n < DSTATE; ++n) {
    Aprod[o + n] = ap[n];
    Hpart[o + n] = hp[n];
  }
}

// ---------------------------------------------------------------------------
// 6b) Scan phase B: prefix over chunk summaries; Hp rewritten in place with
//     the state ENTERING each chunk.  One thread per (bk, d, n).
// ---------------------------------------------------------------------------
__global__ void k_scan_prefix(const float* __restrict__ Aprod,
                              float* __restrict__ Hp)
{
  const long idx = (long)blockIdx.x * blockDim.x + threadIdx.x;
  if (idx >= (long)BATCH * KDIR * DINNER * DSTATE) return;
  const int n = (int)(idx % DSTATE);
  const int d = (int)((idx / DSTATE) % DINNER);
  const int bk = (int)(idx / ((long)DSTATE * DINNER));
  float h = 0.f;
  for (int c = 0; c < SCH; ++c) {
    const long o = (((long)bk * SCH + c) * DINNER + d) * DSTATE + n;
    const float a = Aprod[o];
    const float p = Hp[o];
    Hp[o] = h;               // state entering chunk c
    h = a * h + p;
  }
}

// ---------------------------------------------------------------------------
// 6c) Scan phase C: replay each chunk from its entering state, emit
//     out_y = sum_n h*C + D*u.
// ---------------------------------------------------------------------------
__global__ void k_scan_out(const float* __restrict__ delta,
                           const float* __restrict__ xs,
                           const float* __restrict__ Bsv,
                           const float* __restrict__ Csv,
                           const float* __restrict__ A_logs,
                           const float* __restrict__ Dsv,     // (6*192)
                           const float* __restrict__ Hstart,  // (B,6,SCH,DI,16)
                           float* __restrict__ outy)          // (B,6,L,DI)
{
  const int blk = blockIdx.x;
  const int c  = blk % SCH;
  const int bk = blk / SCH;
  const int k  = bk % KDIR;
  const int d  = threadIdx.x;
  float Arow[DSTATE], h[DSTATE];
  const long o = (((long)bk * SCH + c) * DINNER + d) * DSTATE;
  #pragma unroll
  for (int n = 0; n < DSTATE; ++n) {
    Arow[n] = -__expf(A_logs[((long)k * DINNER + d) * DSTATE + n]);
    h[n] = Hstart[o + n];
  }
  const float Dd = Dsv[k * DINNER + d];
  const int t0 = c * SCLEN;
  const float* dp = delta + (long)bk * LL * DINNER;
  const float* up = xs    + (long)bk * LL * DINNER;
  const float* Bp = Bsv   + (long)bk * LL * DSTATE;
  const float* Cp = Csv   + (long)bk * LL * DSTATE;
  float* yp = outy + (long)bk * LL * DINNER;
  for (int tt = 0; tt < SCLEN; ++tt) {
    const int t = t0 + tt;
    const float dl = dp[(long)t * DINNER + d];
    const float u  = up[(long)t * DINNER + d];
    const float du = dl * u;
    float yacc = 0.f;
    #pragma unroll
    for (int n = 0; n < DSTATE; ++n) {
      const float e = __expf(dl * Arow[n]);
      h[n] = e * h[n] + du * Bp[(long)t * DSTATE + n];
      yacc += h[n] * Cp[(long)t * DSTATE + n];
    }
    yp[(long)t * DINNER + d] = yacc + Dd * u;
  }
}

// ---------------------------------------------------------------------------
// 7) Combine forward/backward scans into y_h, y_v, y_D (B,L,DI).
// ---------------------------------------------------------------------------
__global__ void k_combine(const float* __restrict__ outy,
                          float* __restrict__ yh,
                          float* __restrict__ yv2,
                          float* __restrict__ yD)
{
  const long idx = (long)blockIdx.x * blockDim.x + threadIdx.x;
  if (idx >= (long)BL * DINNER) return;
  const int d = (int)(idx % DINNER);
  const long m = idx / DINNER;
  const int l = (int)(m % LL);
  const int b = (int)(m / LL);
  const long base = (long)b * KDIR * LL * DINNER;
#define OY(kk, pos) outy[base + (long)(kk) * LL * DINNER + (long)(pos) * DINNER + d]
  yh[idx] = OY(0, l) + OY(3, LL - 1 - l);
  const int h = l / WW, w = l % WW;
  const int lv = w * HH + h;
  yv2[idx] = OY(1, lv) + OY(4, LL - 1 - lv);
  const int wq = (w - h + WW) % WW;
  const int lD = wq * HH + h;
  yD[idx] = OY(2, lD) + OY(5, LL - 1 - lD);
#undef OY
}

// ---------------------------------------------------------------------------
// 8) LayerNorm(192) * silu-gate per head -> g (3, BL, DI).
//    One wave per row, butterfly reduce with __shfl_xor (wave32).
// ---------------------------------------------------------------------------
__global__ void k_lngate(const float* __restrict__ yh,
                         const float* __restrict__ yv2,
                         const float* __restrict__ yD,
                         const float* __restrict__ zg,
                         const float* __restrict__ lnw,
                         const float* __restrict__ lnb,
                         float* __restrict__ g)
{
  const long tid = (long)blockIdx.x * blockDim.x + threadIdx.x;
  const long wid = tid >> 5;
  const int lane = (int)(tid & 31);
  if (wid >= (long)3 * BL) return;
  const int m = (int)(wid % BL);
  const int j = (int)(wid / BL);
  const float* src = (j == 0 ? yh : (j == 1 ? yv2 : yD)) + (long)m * DINNER;
  float s = 0.f, ss = 0.f;
  #pragma unroll
  for (int dd = lane; dd < DINNER; dd += 32) {
    const float v = src[dd];
    s += v; ss += v * v;
  }
  #pragma unroll
  for (int off = 16; off > 0; off >>= 1) {
    s  += __shfl_xor(s,  off, 32);
    ss += __shfl_xor(ss, off, 32);
  }
  const float mu   = s * (1.f / (float)DINNER);
  const float var  = ss * (1.f / (float)DINNER) - mu * mu;
  const float rstd = rsqrtf(var + 1e-5f);
  float* gp = g + ((long)j * BL + m) * DINNER;
  const float* zp = zg + (long)m * DINNER;
  #pragma unroll
  for (int dd = lane; dd < DINNER; dd += 32) {
    const float t = (src[dd] - mu) * rstd * lnw[dd] + lnb[dd];
    gp[dd] = t * zp[dd];
  }
}

// ---------------------------------------------------------------------------
// 9) out_proj GEMM for 3 heads -> d_out.
// ---------------------------------------------------------------------------
__global__ void k_outproj(const float* __restrict__ g,   // (3, BL, 192)
                          const float* __restrict__ Wo,  // (96, 192)
                          float* __restrict__ outp)      // (3, BL, 96)
{
  const int MT = BL / 16;       // 288
  const int NT = DMODEL / 16;   // 6
  const int tile = blockIdx.x;
  const int j = tile / (MT * NT);
  const int rem = tile % (MT * NT);
  const int mtile = rem / NT;
  const int ntile = rem % NT;
  const float* Arow = g + ((long)j * BL + mtile * 16) * DINNER;
  const float* Wrow = Wo + (long)ntile * 16 * DINNER;
  v8f acc = wave_gemm_tile<DINNER>(Arow, DINNER, Wrow, DINNER, 16);
  const int lane = threadIdx.x & 31;
  const int lhalf = lane >> 4, ncol = lane & 15;
  #pragma unroll
  for (int r = 0; r < 8; ++r) {
    const int m = mtile * 16 + r + lhalf * 8;
    const int c = ntile * 16 + ncol;
    outp[((long)j * BL + m) * DMODEL + c] = acc[r];
  }
}

// ---------------------------------------------------------------------------
extern "C" void kernel_launch(void* const* d_in, const int* in_sizes, int n_in,
                              void* d_out, int out_size, void* d_ws, size_t ws_size,
                              hipStream_t stream)
{
  (void)in_sizes; (void)n_in; (void)out_size; (void)ws_size;
  const float* x    = (const float*)d_in[0];
  const float* y    = (const float*)d_in[1];
  const float* kin  = (const float*)d_in[2];
  const float* inw  = (const float*)d_in[3];
  const float* cw   = (const float*)d_in[4];
  const float* cb   = (const float*)d_in[5];
  const float* xpw  = (const float*)d_in[6];
  const float* dtw  = (const float*)d_in[7];
  const float* dtb  = (const float*)d_in[8];
  const float* alog = (const float*)d_in[9];
  const float* dsv  = (const float*)d_in[10];
  const float* lnw  = (const float*)d_in[11];
  const float* lnb  = (const float*)d_in[12];
  const float* ow   = (const float*)d_in[13];
  float* outp = (float*)d_out;
  float* ws   = (float*)d_ws;

  const long SZ    = (long)BL * DINNER;                        // 884,736
  const long SZX   = (long)BATCH * KDIR * LL * DINNER;         // 5,308,416
  const long SZCH  = (long)BATCH * KDIR * SCH * DINNER * DSTATE; // 1,179,648
  float* xv    = ws;            // reused later as y_h
  float* yv    = xv + SZ;       // reused later as y_v
  float* zg    = yv + SZ;
  float* xc    = zg + SZ;       // reused later as y_D
  float* yc    = xc + SZ;
  float* xs    = yc + SZ;       // reused later as g (3*SZ <= SZX)
  float* delta = xs + SZX;
  float* dtr   = delta + SZX;
  float* Bsv   = dtr + (long)BATCH * KDIR * LL * DTRANK;
  float* Csv   = Bsv + (long)BATCH * KDIR * LL * DSTATE;
  float* outy  = Csv + (long)BATCH * KDIR * LL * DSTATE;
  float* Aprod = outy + SZX;
  float* Hpart = Aprod + SZCH;  // phase B rewrites in place as Hstart

  const dim3 w32(32);
  const int tb = 256;

  // 1) input projections (WMMA)
  const int tiles_in = (BL / 16) * (DINNER / 16);  // 3456
  k_inproj<<<tiles_in, w32, 0, stream>>>(x,   inw, xv, 0,      0);
  k_inproj<<<tiles_in, w32, 0, stream>>>(y,   inw, yv, 0,      0);
  k_inproj<<<tiles_in, w32, 0, stream>>>(kin, inw, zg, DINNER, 1);

  // 2) depthwise conv + SiLU
  k_conv3x3_silu<<<(int)((SZ + tb - 1) / tb), tb, 0, stream>>>(xv, cw, cb, xc);
  k_conv3x3_silu<<<(int)((SZ + tb - 1) / tb), tb, 0, stream>>>(yv, cw, cb, yc);

  // 3) 6 directional sequences
  k_build_xs<<<(int)((SZX + tb - 1) / tb), tb, 0, stream>>>(xc, yc, xs);

  // 4) x_proj (WMMA, N padded 38->48)
  const int tiles_xd = KDIR * (BL / 16) * 3;  // 5184
  k_xdbl<<<tiles_xd, w32, 0, stream>>>(xs, xpw, dtr, Bsv, Csv);

  // 5) delta
  k_delta<<<(int)((SZX + tb - 1) / tb), tb, 0, stream>>>(dtr, dtw, dtb, delta);

  // 6) chunk-parallel selective scan: A) chunk summaries, B) prefix, C) replay
  k_scan_chunks<<<BATCH * KDIR * SCH, DINNER, 0, stream>>>(
      delta, xs, Bsv, alog, Aprod, Hpart);
  const long npre = (long)BATCH * KDIR * DINNER * DSTATE;  // 36,864
  k_scan_prefix<<<(int)((npre + tb - 1) / tb), tb, 0, stream>>>(Aprod, Hpart);
  k_scan_out<<<BATCH * KDIR * SCH, DINNER, 0, stream>>>(
      delta, xs, Bsv, Csv, alog, dsv, Hpart, outy);

  // 7) combine forward/backward scans (writes into dead xv/yv/xc regions)
  k_combine<<<(int)((SZ + tb - 1) / tb), tb, 0, stream>>>(outy, xv, yv, xc);

  // 8) LayerNorm * gate -> g (reuses dead xs region); one wave per row
  float* g = xs;
  const long nln = (long)3 * BL * 32;  // threads: one wave per row
  k_lngate<<<(int)((nln + tb - 1) / tb), tb, 0, stream>>>(
      xv, yv, xc, zg, lnw, lnb, g);

  // 9) out_proj (WMMA) -> three heads concatenated in d_out
  const int tiles_out = 3 * (BL / 16) * (DMODEL / 16);  // 5184
  k_outproj<<<tiles_out, w32, 0, stream>>>(g, ow, outp);
}
